// BasicTransformerBlock_81570018885849
// MI455X (gfx1250) — compile-verified, matrix operand order checked by reading, not run on hardware
//
#include <hip/hip_runtime.h>
#include <hip/hip_bf16.h>

// ---------------- constants (from reference) ----------------
#define DIM   1280
#define NH    8
#define DH    160        // head dim
#define DC    768
#define NF    16         // frames (video_length)
#define SEQ   1024
#define ENC   93
#define IPTK  16
#define ENDP  77         // ENC - IPTK
#define LTXT  96         // 77 padded to mult of 32
#define LIMG  32         // 16 padded to mult of 32
#define FFD   5120
#define FF2   10240
#define MTOT  (NF * SEQ) // 16384
#define FFCH  2048       // FF row chunk
#define SCALE_ATTN 0.07905694150420949f // 160^-0.5

#define DHC 5            // DH / 32  (k-chunks per head)
#define DNT 10           // DH / 16  (output col tiles per head)

typedef __attribute__((ext_vector_type(16))) __bf16 v16bf;
typedef __attribute__((ext_vector_type(8)))  __bf16 v8bf;
typedef __attribute__((ext_vector_type(8)))  float  v8f;
typedef __attribute__((ext_vector_type(4)))  int    v4i;

union Frag16 { v16bf v; v8bf h[2]; };

static __device__ __forceinline__ v8f wmma_bf16(const v16bf& a, const v16bf& b, const v8f& c) {
    return __builtin_amdgcn_wmma_f32_16x16x32_bf16(false, a, false, b, (short)0, c, false, false);
}

// ---------------- CDNA5 async global->LDS copy (16B per lane) ----------------
#define AS1 __attribute__((address_space(1)))
#define AS3 __attribute__((address_space(3)))

#if __has_builtin(__builtin_amdgcn_global_load_async_to_lds_b128) && \
    __has_builtin(__builtin_amdgcn_s_wait_asynccnt)
#define USE_ASYNC_LDS 1
#define ASYNC_COPY16(dst, src)                                                  \
    __builtin_amdgcn_global_load_async_to_lds_b128(                             \
        (AS1 v4i*)(size_t)(const void*)(src), (AS3 v4i*)(dst), 0, 0)
#define WAIT_ASYNC(n) __builtin_amdgcn_s_wait_asynccnt(n)
#elif defined(__gfx1250__)
#define USE_ASYNC_LDS 1
#define ASYNC_COPY16(dst, src)                                                  \
    asm volatile("global_load_async_to_lds_b128 %0, %1, off"                    \
                 :: "v"((unsigned)(size_t)(AS3 void*)(dst)),                    \
                    "v"((const void*)(src)) : "memory")
#define WAIT_ASYNC(n) asm volatile("s_wait_asynccnt " #n ::: "memory")
#else
#define USE_ASYNC_LDS 0
#define ASYNC_COPY16(dst, src) (*(v8bf*)(dst) = *(const v8bf*)(src))
#define WAIT_ASYNC(n)
#endif

// ---------------- LayerNorm: f32 row -> bf16 row ----------------
__global__ __launch_bounds__(256)
void ln_kernel(const float* __restrict__ x, const float* __restrict__ g,
               const float* __restrict__ b, __bf16* __restrict__ y) {
    __shared__ float s1[256], s2[256];
    const float* xr = x + (size_t)blockIdx.x * DIM;
    float a0 = 0.f, a1 = 0.f;
    for (int i = threadIdx.x; i < DIM; i += 256) { float v = xr[i]; a0 += v; a1 += v * v; }
    s1[threadIdx.x] = a0; s2[threadIdx.x] = a1;
    __syncthreads();
    for (int off = 128; off > 0; off >>= 1) {
        if ((int)threadIdx.x < off) {
            s1[threadIdx.x] += s1[threadIdx.x + off];
            s2[threadIdx.x] += s2[threadIdx.x + off];
        }
        __syncthreads();
    }
    float mean = s1[0] * (1.0f / DIM);
    float var  = s2[0] * (1.0f / DIM) - mean * mean;
    float rstd = rsqrtf(var + 1e-5f);
    __bf16* yr = y + (size_t)blockIdx.x * DIM;
    for (int i = threadIdx.x; i < DIM; i += 256)
        yr[i] = (__bf16)((xr[i] - mean) * rstd * g[i] + b[i]);
}

// ---------------- weight prep: f32 [K][N] -> bf16 [N][K] ----------------
__global__ __launch_bounds__(256)
void wt_prep(const float* __restrict__ W, __bf16* __restrict__ Wt, int K, int N) {
    size_t idx = (size_t)blockIdx.x * 256 + threadIdx.x;
    size_t tot = (size_t)K * N;
    if (idx >= tot) return;
    int k = (int)(idx / N), n = (int)(idx % N);
    Wt[(size_t)n * K + k] = (__bf16)W[idx];
}

// ---------------- encoder pad: f32 [16][93][768] slice -> bf16 [16][Lpad][768] ----------------
__global__ __launch_bounds__(256)
void pad_enc(const float* __restrict__ enc, __bf16* __restrict__ out,
             int Lsrc, int Lpad, int off) {
    size_t idx = (size_t)blockIdx.x * 256 + threadIdx.x;
    size_t tot = (size_t)NF * Lpad * DC;
    if (idx >= tot) return;
    int c = (int)(idx % DC);
    int s = (int)((idx / DC) % Lpad);
    int f = (int)(idx / ((size_t)DC * Lpad));
    float v = (s < Lsrc) ? enc[((size_t)f * ENC + off + s) * DC + c] : 0.f;
    out[idx] = (__bf16)v;
}

// ---------------- V transpose: bf16 [16][Sv][1280] -> bf16 [16][8][160][Sv] ----------------
__global__ __launch_bounds__(256)
void vtrans(const __bf16* __restrict__ v, __bf16* __restrict__ vt, int Sv) {
    size_t idx = (size_t)blockIdx.x * 256 + threadIdx.x;
    size_t tot = (size_t)NF * Sv * DIM;
    if (idx >= tot) return;
    int col = (int)(idx % DIM);
    int s   = (int)((idx / DIM) % Sv);
    int f   = (int)(idx / ((size_t)DIM * Sv));
    int h = col / DH, d = col % DH;
    vt[(((size_t)f * NH + h) * DH + d) * Sv + s] = v[idx];
}

// ---------------- bf16 + bf16 -> bf16 ----------------
__global__ __launch_bounds__(256)
void sum2_bf16(const __bf16* __restrict__ a, const __bf16* __restrict__ b,
               __bf16* __restrict__ o, size_t n) {
    size_t idx = (size_t)blockIdx.x * 256 + threadIdx.x;
    if (idx >= n) return;
    o[idx] = (__bf16)((float)a[idx] + (float)b[idx]);
}

// ---------------- GEGLU: proj f32 [rows][10240] -> bf16 [rows][5120] ----------------
__global__ __launch_bounds__(256)
void geglu(const float* __restrict__ proj, __bf16* __restrict__ t, int rows) {
    size_t idx = (size_t)blockIdx.x * 256 + threadIdx.x;
    size_t tot = (size_t)rows * FFD;
    if (idx >= tot) return;
    int r = (int)(idx / FFD), c = (int)(idx % FFD);
    float a = proj[(size_t)r * FF2 + c];
    float g = proj[(size_t)r * FF2 + FFD + c];
    float ge = 0.5f * g * (1.f + erff(g * 0.70710678118654752f));
    t[idx] = (__bf16)(a * ge);
}

// ---------------- WMMA GEMM with async-LDS double buffering ----------------
// C[M][N] = A[M][K](bf16, row-major) x Wt[N][K](bf16, pre-transposed)
// Block: 128 threads (4 waves). Tile: 64(M) x 64(N), K stepped by 32.
// Stage: 64x32 A panel + 64x32 B panel -> LDS via GLOBAL_LOAD_ASYNC_TO_LDS_B128
// (4 async ops / thread / stage), double buffered on ASYNCcnt.
// mode 0: Cf = acc + bias ; mode 1: Cb = bf16(acc + bias) ; mode 2: Cadd += acc + bias
#define GM 64
#define GN 64
#define GK 32
#define GKP 40   // padded LDS row stride (80B) to spread across the 64 banks

__global__ __launch_bounds__(128)
void gemm_bf16(const __bf16* __restrict__ A, const __bf16* __restrict__ Bt,
               const float* __restrict__ bias,
               float* __restrict__ Cf, __bf16* __restrict__ Cb, float* __restrict__ Cadd,
               int M, int N, int K, int mode) {
    __shared__ __bf16 sA[2][GM * GKP];
    __shared__ __bf16 sB[2][GN * GKP];
    const int tid  = threadIdx.x;
    const int lane = tid & 31;
    const int wave = tid >> 5;
    const int half = lane >> 4;
    const int n    = lane & 15;
    const int mBase = blockIdx.x * GM;
    const int nBase = blockIdx.y * GN;

    auto stage = [&](int buf, int k0) {
#pragma unroll
        for (int i = 0; i < 2; ++i) {
            int cidx = tid * 2 + i;            // 0..255 : 64 rows x 4 x 16B chunks
            int r = cidx >> 2, c = cidx & 3;
            const __bf16* ga = A  + (size_t)(mBase + r) * K + k0 + c * 8;
            const __bf16* gb = Bt + (size_t)(nBase + r) * K + k0 + c * 8;
            ASYNC_COPY16(&sA[buf][r * GKP + c * 8], ga);
            ASYNC_COPY16(&sB[buf][r * GKP + c * 8], gb);
        }
    };

    const int steps = K / GK;
    v8f acc[4] = {};
    stage(0, 0);
    for (int it = 0; it < steps; ++it) {
        const int cur = it & 1;
        if (it + 1 < steps) {
            stage(cur ^ 1, (it + 1) * GK);
            WAIT_ASYNC(4);   // retire previous stage; newest stays in flight
        } else {
            WAIT_ASYNC(0);   // drain before consuming the final panel
        }
        __syncthreads();

        Frag16 a;
        const __bf16* ar = &sA[cur][(wave * 16 + n) * GKP];
        a.h[0] = *(const v8bf*)(ar + half * 8);
        a.h[1] = *(const v8bf*)(ar + 16 + half * 8);
#pragma unroll
        for (int t = 0; t < 4; ++t) {
            const __bf16* br = &sB[cur][(t * 16 + n) * GKP];
            Frag16 b;
            b.h[0] = *(const v8bf*)(br + half * 8);
            b.h[1] = *(const v8bf*)(br + 16 + half * 8);
            acc[t] = wmma_bf16(a.v, b.v, acc[t]);
        }
        __syncthreads();
    }

#pragma unroll
    for (int t = 0; t < 4; ++t)
#pragma unroll
        for (int r = 0; r < 8; ++r) {
            int row = mBase + wave * 16 + r + 8 * half;
            int col = nBase + t * 16 + n;
            float v = acc[t][r] + (bias ? bias[col] : 0.f);
            size_t idx = (size_t)row * N + col;
            if (mode == 0)      Cf[idx] = v;
            else if (mode == 1) Cb[idx] = (__bf16)v;
            else                Cadd[idx] += v;
        }
}

// ---------------- Flash attention, one wave per (qtile, head, frame) ----------------
// q: bf16 [NF][Sq][DIM]; k: bf16 [NF][Lpad][DIM]; vt: bf16 [NF][NH][DH][Lpad]
// out: bf16 [NF][Sq][DIM]. bcast: use frame 0 for k/vt.
__global__ __launch_bounds__(32)
void attn_kernel(const __bf16* __restrict__ q, const __bf16* __restrict__ k,
                 const __bf16* __restrict__ vt, __bf16* __restrict__ out,
                 int Sq, int Lpad, int Lvalid, int bcast) {
    __shared__ __bf16 pbuf[16 * 32];
    const int lane = threadIdx.x;
    const int half = lane >> 4;
    const int n    = lane & 15;
    const int qt = blockIdx.x;
    const int h  = blockIdx.y;
    const int f  = blockIdx.z;
    const int fk = bcast ? 0 : f;

    const __bf16* qrow = q + ((size_t)f * Sq + qt * 16 + n) * DIM + h * DH;
    v16bf qa[DHC];
#pragma unroll
    for (int c = 0; c < DHC; ++c) {
        Frag16 t;
        t.h[0] = *(const v8bf*)(qrow + c * 32 + half * 8);
        t.h[1] = *(const v8bf*)(qrow + c * 32 + 16 + half * 8);
        qa[c] = t.v;
    }

    v8f oacc[DNT] = {};
    float mrow[8], srow[8];
#pragma unroll
    for (int r = 0; r < 8; ++r) { mrow[r] = -1e30f; srow[r] = 0.f; }

    const __bf16* kbase = k + (size_t)fk * Lpad * DIM + h * DH;
    const __bf16* vbase = vt + ((size_t)fk * NH + h) * (size_t)DH * Lpad;

    for (int kb = 0; kb < Lpad; kb += 32) {
        v8f s0 = {}, s1 = {};
        const __bf16* kr0 = kbase + (size_t)(kb + n) * DIM;
        const __bf16* kr1 = kbase + (size_t)(kb + 16 + n) * DIM;
#pragma unroll
        for (int c = 0; c < DHC; ++c) {
            Frag16 b0, b1;
            b0.h[0] = *(const v8bf*)(kr0 + c * 32 + half * 8);
            b0.h[1] = *(const v8bf*)(kr0 + c * 32 + 16 + half * 8);
            b1.h[0] = *(const v8bf*)(kr1 + c * 32 + half * 8);
            b1.h[1] = *(const v8bf*)(kr1 + c * 32 + 16 + half * 8);
            s0 = wmma_bf16(qa[c], b0.v, s0);
            s1 = wmma_bf16(qa[c], b1.v, s1);
        }
        const bool msk0 = (kb + n)      >= Lvalid;
        const bool msk1 = (kb + 16 + n) >= Lvalid;
#pragma unroll
        for (int r = 0; r < 8; ++r) {
            s0[r] = msk0 ? -1e30f : s0[r] * SCALE_ATTN;
            s1[r] = msk1 ? -1e30f : s1[r] * SCALE_ATTN;
        }
        float corr[8];
#pragma unroll
        for (int r = 0; r < 8; ++r) {
            float rm = fmaxf(s0[r], s1[r]);
            rm = fmaxf(rm, __shfl_xor(rm, 1, 16));
            rm = fmaxf(rm, __shfl_xor(rm, 2, 16));
            rm = fmaxf(rm, __shfl_xor(rm, 4, 16));
            rm = fmaxf(rm, __shfl_xor(rm, 8, 16));
            float nm = fmaxf(mrow[r], rm);
            corr[r] = __expf(mrow[r] - nm);
            float p0 = __expf(s0[r] - nm);
            float p1 = __expf(s1[r] - nm);
            float rs = p0 + p1;
            rs += __shfl_xor(rs, 1, 16);
            rs += __shfl_xor(rs, 2, 16);
            rs += __shfl_xor(rs, 4, 16);
            rs += __shfl_xor(rs, 8, 16);
            srow[r] = srow[r] * corr[r] + rs;
            mrow[r] = nm;
            s0[r] = p0; s1[r] = p1;
        }
#pragma unroll
        for (int c = 0; c < DNT; ++c)
#pragma unroll
            for (int r = 0; r < 8; ++r) oacc[c][r] *= corr[r];

        // stage P to LDS: C-layout -> A-layout
        __syncthreads();
#pragma unroll
        for (int r = 0; r < 8; ++r) {
            int row = r + 8 * half;
            pbuf[row * 32 + n]      = (__bf16)s0[r];
            pbuf[row * 32 + 16 + n] = (__bf16)s1[r];
        }
        __syncthreads();
        Frag16 pa;
        pa.h[0] = *(const v8bf*)(&pbuf[n * 32 + half * 8]);
        pa.h[1] = *(const v8bf*)(&pbuf[n * 32 + 16 + half * 8]);

        // O += P x V (V^T rows contiguous along keys)
#pragma unroll
        for (int c = 0; c < DNT; ++c) {
            const __bf16* vr = vbase + (size_t)(c * 16 + n) * Lpad + kb;
            Frag16 vb;
            vb.h[0] = *(const v8bf*)(vr + half * 8);
            vb.h[1] = *(const v8bf*)(vr + 16 + half * 8);
            oacc[c] = wmma_bf16(pa.v, vb.v, oacc[c]);
        }
    }
#pragma unroll
    for (int c = 0; c < DNT; ++c)
#pragma unroll
        for (int r = 0; r < 8; ++r) {
            int row = qt * 16 + r + 8 * half;
            out[((size_t)f * Sq + row) * DIM + h * DH + c * 16 + n] =
                (__bf16)(oacc[c][r] / srow[r]);
        }
}

// ======================= host orchestration =======================
static inline size_t align256(size_t x) { return (x + 255) & ~(size_t)255; }

extern "C" void kernel_launch(void* const* d_in, const int* in_sizes, int n_in,
                              void* d_out, int out_size, void* d_ws, size_t ws_size,
                              hipStream_t stream) {
    (void)in_sizes; (void)n_in; (void)ws_size;
    const float* h_in  = (const float*)d_in[0];
    const float* enc   = (const float*)d_in[1];
    const float* ln1_g = (const float*)d_in[3];
    const float* ln1_b = (const float*)d_in[4];
    const float* ln2_g = (const float*)d_in[5];
    const float* ln2_b = (const float*)d_in[6];
    const float* ln3_g = (const float*)d_in[7];
    const float* ln3_b = (const float*)d_in[8];
    const float* a1_wq    = (const float*)d_in[9];
    const float* a1_wq_ff = (const float*)d_in[10];
    const float* a1_wk    = (const float*)d_in[11];
    const float* a1_wv    = (const float*)d_in[12];
    const float* a1_wo    = (const float*)d_in[13];
    const float* a1_wo_b  = (const float*)d_in[14];
    const float* a1_wo_ff   = (const float*)d_in[15];
    const float* a1_wo_ff_b = (const float*)d_in[16];
    const float* a2_wq    = (const float*)d_in[17];
    const float* a2_wk    = (const float*)d_in[18];
    const float* a2_wv    = (const float*)d_in[19];
    const float* a2_wk_ip = (const float*)d_in[20];
    const float* a2_wv_ip = (const float*)d_in[21];
    const float* a2_wo    = (const float*)d_in[22];
    const float* a2_wo_b  = (const float*)d_in[23];
    const float* ff_w1 = (const float*)d_in[24];
    const float* ff_b1 = (const float*)d_in[25];
    const float* ff_w2 = (const float*)d_in[26];
    const float* ff_b2 = (const float*)d_in[27];

    // ---- workspace bump allocator ----
    char* wp = (char*)d_ws;
    auto alloc = [&](size_t bytes) -> void* { void* r = (void*)wp; wp += align256(bytes); return r; };
    const size_t MD = (size_t)MTOT * DIM;

    float*  h_cur = (float*)alloc(MD * 4);
    __bf16* nb    = (__bf16*)alloc(MD * 2);
    __bf16* qb    = (__bf16*)alloc(MD * 2);
    __bf16* kb    = (__bf16*)alloc(MD * 2);
    __bf16* vb    = (__bf16*)alloc(MD * 2);
    __bf16* qfb   = (__bf16*)alloc(MD * 2);
    __bf16* vtb   = (__bf16*)alloc(MD * 2);
    __bf16* o1b   = (__bf16*)alloc(MD * 2);
    __bf16* o1fb  = (__bf16*)alloc(MD * 2);
    __bf16* q2b   = (__bf16*)alloc(MD * 2);
    __bf16* otxtb = (__bf16*)alloc(MD * 2);
    __bf16* oimgb = (__bf16*)alloc(MD * 2);
    __bf16* o2b   = (__bf16*)alloc(MD * 2);

    __bf16* txtp  = (__bf16*)alloc((size_t)NF * LTXT * DC * 2);
    __bf16* imgp  = (__bf16*)alloc((size_t)NF * LIMG * DC * 2);
    __bf16* ktxt  = (__bf16*)alloc((size_t)NF * LTXT * DIM * 2);
    __bf16* vtxt  = (__bf16*)alloc((size_t)NF * LTXT * DIM * 2);
    __bf16* vttxt = (__bf16*)alloc((size_t)NF * LTXT * DIM * 2);
    __bf16* kimg  = (__bf16*)alloc((size_t)NF * LIMG * DIM * 2);
    __bf16* vimg  = (__bf16*)alloc((size_t)NF * LIMG * DIM * 2);
    __bf16* vtimg = (__bf16*)alloc((size_t)NF * LIMG * DIM * 2);

    const size_t WDD = (size_t)DIM * DIM;
    __bf16* wq_t    = (__bf16*)alloc(WDD * 2);
    __bf16* wqff_t  = (__bf16*)alloc(WDD * 2);
    __bf16* wk_t    = (__bf16*)alloc(WDD * 2);
    __bf16* wv_t    = (__bf16*)alloc(WDD * 2);
    __bf16* wo_t    = (__bf16*)alloc(WDD * 2);
    __bf16* woff_t  = (__bf16*)alloc(WDD * 2);
    __bf16* w2q_t   = (__bf16*)alloc(WDD * 2);
    __bf16* w2o_t   = (__bf16*)alloc(WDD * 2);
    __bf16* w2k_t   = (__bf16*)alloc((size_t)DC * DIM * 2);
    __bf16* w2v_t   = (__bf16*)alloc((size_t)DC * DIM * 2);
    __bf16* w2kip_t = (__bf16*)alloc((size_t)DC * DIM * 2);
    __bf16* w2vip_t = (__bf16*)alloc((size_t)DC * DIM * 2);
    __bf16* w1_t    = (__bf16*)alloc((size_t)DIM * FF2 * 2);
    __bf16* w2_t    = (__bf16*)alloc((size_t)FFD * DIM * 2);

    float*  proj = (float*)alloc((size_t)FFCH * FF2 * 4);
    __bf16* tch  = (__bf16*)alloc((size_t)FFCH * FFD * 2);

    auto eb = [](size_t n) { return (unsigned)((n + 255) / 256); };

    // 0. residual stream
    (void)hipMemcpyAsync(h_cur, h_in, MD * 4, hipMemcpyDeviceToDevice, stream);

    // 1. weight prep (f32 [K][N] -> bf16 [N][K])
    wt_prep<<<eb(WDD), 256, 0, stream>>>(a1_wq,    wq_t,    DIM, DIM);
    wt_prep<<<eb(WDD), 256, 0, stream>>>(a1_wq_ff, wqff_t,  DIM, DIM);
    wt_prep<<<eb(WDD), 256, 0, stream>>>(a1_wk,    wk_t,    DIM, DIM);
    wt_prep<<<eb(WDD), 256, 0, stream>>>(a1_wv,    wv_t,    DIM, DIM);
    wt_prep<<<eb(WDD), 256, 0, stream>>>(a1_wo,    wo_t,    DIM, DIM);
    wt_prep<<<eb(WDD), 256, 0, stream>>>(a1_wo_ff, woff_t,  DIM, DIM);
    wt_prep<<<eb(WDD), 256, 0, stream>>>(a2_wq,    w2q_t,   DIM, DIM);
    wt_prep<<<eb(WDD), 256, 0, stream>>>(a2_wo,    w2o_t,   DIM, DIM);
    wt_prep<<<eb((size_t)DC * DIM), 256, 0, stream>>>(a2_wk,    w2k_t,   DC, DIM);
    wt_prep<<<eb((size_t)DC * DIM), 256, 0, stream>>>(a2_wv,    w2v_t,   DC, DIM);
    wt_prep<<<eb((size_t)DC * DIM), 256, 0, stream>>>(a2_wk_ip, w2kip_t, DC, DIM);
    wt_prep<<<eb((size_t)DC * DIM), 256, 0, stream>>>(a2_wv_ip, w2vip_t, DC, DIM);
    wt_prep<<<eb((size_t)DIM * FF2), 256, 0, stream>>>(ff_w1, w1_t, DIM, FF2);
    wt_prep<<<eb((size_t)FFD * DIM), 256, 0, stream>>>(ff_w2, w2_t, FFD, DIM);

    // 2. stage 1: self-attention (temporal-frame variants)
    ln_kernel<<<MTOT, 256, 0, stream>>>(h_cur, ln1_g, ln1_b, nb);
    dim3 gQ(MTOT / GM, DIM / GN);
    gemm_bf16<<<gQ, 128, 0, stream>>>(nb, wq_t,   nullptr, nullptr, qb,  nullptr, MTOT, DIM, DIM, 1);
    gemm_bf16<<<gQ, 128, 0, stream>>>(nb, wqff_t, nullptr, nullptr, qfb, nullptr, MTOT, DIM, DIM, 1);
    gemm_bf16<<<gQ, 128, 0, stream>>>(nb, wk_t,   nullptr, nullptr, kb,  nullptr, MTOT, DIM, DIM, 1);
    gemm_bf16<<<gQ, 128, 0, stream>>>(nb, wv_t,   nullptr, nullptr, vb,  nullptr, MTOT, DIM, DIM, 1);
    vtrans<<<eb(MD), 256, 0, stream>>>(vb, vtb, SEQ);

    dim3 gA(SEQ / 16, NH, NF);
    attn_kernel<<<gA, 32, 0, stream>>>(qb,  kb, vtb, o1b,  SEQ, SEQ, SEQ, 0);
    attn_kernel<<<gA, 32, 0, stream>>>(qfb, kb, vtb, o1fb, SEQ, SEQ, SEQ, 1); // k_ff/v_ff = frame 0

    gemm_bf16<<<gQ, 128, 0, stream>>>(o1b,  wo_t,   a1_wo_b,    nullptr, nullptr, h_cur, MTOT, DIM, DIM, 2);
    gemm_bf16<<<gQ, 128, 0, stream>>>(o1fb, woff_t, a1_wo_ff_b, nullptr, nullptr, h_cur, MTOT, DIM, DIM, 2);

    // 3. stage 2: cross-attention (txt + img adapter)
    ln_kernel<<<MTOT, 256, 0, stream>>>(h_cur, ln2_g, ln2_b, nb);
    gemm_bf16<<<gQ, 128, 0, stream>>>(nb, w2q_t, nullptr, nullptr, q2b, nullptr, MTOT, DIM, DIM, 1);

    pad_enc<<<eb((size_t)NF * LTXT * DC), 256, 0, stream>>>(enc, txtp, ENDP, LTXT, 0);
    pad_enc<<<eb((size_t)NF * LIMG * DC), 256, 0, stream>>>(enc, imgp, IPTK, LIMG, ENDP);

    dim3 gT(NF * LTXT / GM, DIM / GN);
    gemm_bf16<<<gT, 128, 0, stream>>>(txtp, w2k_t, nullptr, nullptr, ktxt, nullptr, NF * LTXT, DIM, DC, 1);
    gemm_bf16<<<gT, 128, 0, stream>>>(txtp, w2v_t, nullptr, nullptr, vtxt, nullptr, NF * LTXT, DIM, DC, 1);
    vtrans<<<eb((size_t)NF * LTXT * DIM), 256, 0, stream>>>(vtxt, vttxt, LTXT);

    dim3 gI(NF * LIMG / GM, DIM / GN);
    gemm_bf16<<<gI, 128, 0, stream>>>(imgp, w2kip_t, nullptr, nullptr, kimg, nullptr, NF * LIMG, DIM, DC, 1);
    gemm_bf16<<<gI, 128, 0, stream>>>(imgp, w2vip_t, nullptr, nullptr, vimg, nullptr, NF * LIMG, DIM, DC, 1);
    vtrans<<<eb((size_t)NF * LIMG * DIM), 256, 0, stream>>>(vimg, vtimg, LIMG);

    attn_kernel<<<gA, 32, 0, stream>>>(q2b, ktxt, vttxt, otxtb, SEQ, LTXT, ENDP, 0);
    attn_kernel<<<gA, 32, 0, stream>>>(q2b, kimg, vtimg, oimgb, SEQ, LIMG, IPTK, 0);
    sum2_bf16<<<eb(MD), 256, 0, stream>>>(otxtb, oimgb, o2b, MD);

    gemm_bf16<<<gQ, 128, 0, stream>>>(o2b, w2o_t, a2_wo_b, nullptr, nullptr, h_cur, MTOT, DIM, DIM, 2);

    // 4. stage 3: GEGLU feed-forward (row-chunked to bound workspace)
    ln_kernel<<<MTOT, 256, 0, stream>>>(h_cur, ln3_g, ln3_b, nb);
    for (int ch = 0; ch < MTOT / FFCH; ++ch) {
        const __bf16* a_ch = nb + (size_t)ch * FFCH * DIM;
        float* h_ch = h_cur + (size_t)ch * FFCH * DIM;
        dim3 g1(FFCH / GM, FF2 / GN);
        gemm_bf16<<<g1, 128, 0, stream>>>(a_ch, w1_t, ff_b1, proj, nullptr, nullptr, FFCH, FF2, DIM, 0);
        geglu<<<eb((size_t)FFCH * FFD), 256, 0, stream>>>(proj, tch, FFCH);
        dim3 g2(FFCH / GM, DIM / GN);
        gemm_bf16<<<g2, 128, 0, stream>>>(tch, w2_t, ff_b2, nullptr, nullptr, h_ch, FFCH, DIM, FFD, 2);
    }

    // 5. output
    (void)hipMemcpyAsync(d_out, h_cur, (size_t)out_size * 4, hipMemcpyDeviceToDevice, stream);
}
